// GATModel_16037407883541
// MI455X (gfx1250) — compile-verified, matrix operand order checked by reading, not run on hardware
//
#include <hip/hip_runtime.h>
#include <hip/hip_bf16.h>

typedef __attribute__((ext_vector_type(16))) _Float16 v16h;
typedef __attribute__((ext_vector_type(8)))  float    v8f;
typedef __attribute__((ext_vector_type(4)))  float    f32x4;
typedef __attribute__((ext_vector_type(2)))  float    f32x2;

#define NNODES   50000
#define NEDGES   800000
#define NETOT    (NEDGES + NNODES)   // with self loops
#define CIN      128
#define NHEADS   4
#define CHID     32
#define COUT     64
#define BN_EPS   1e-5f
#define LRELU_S  0.2f

typedef long long ei_t;   // edge_index dtype is jnp.int64 in the reference

template<int N> struct VecOf;
template<> struct VecOf<4> { typedef f32x4 type; };
template<> struct VecOf<2> { typedef f32x2 type; };

// ---------------------------------------------------------------------------
// utility kernels
// ---------------------------------------------------------------------------
__global__ void fill_f32_kernel(float* __restrict__ p, float v, int n) {
    int i = blockIdx.x * blockDim.x + threadIdx.x;
    if (i < n) p[i] = v;
}

__global__ void add_bias_kernel(float* __restrict__ x, const float* __restrict__ bias,
                                int rows, int C) {
    int i = blockIdx.x * blockDim.x + threadIdx.x;
    if (i < rows * C) x[i] += bias[i % C];
}

// per-channel sum & sum-of-squares (for BN). sums has 2*C floats, pre-zeroed.
__global__ void bn_stats_kernel(const float* __restrict__ x, float* __restrict__ sums,
                                int rows, int C) {
    int c = threadIdx.x;              // blockDim.x == C
    float s = 0.f, ss = 0.f;
    for (int r = blockIdx.x; r < rows; r += gridDim.x) {
        float v = x[(size_t)r * C + c];
        s += v; ss += v * v;
    }
    atomicAdd(&sums[c], s);
    atomicAdd(&sums[C + c], ss);
}

// y = gamma*(x-mean)*rsqrt(var+eps)+beta ; optional ELU ; optional +residual
__global__ void bn_apply_kernel(const float* __restrict__ x, const float* __restrict__ sums,
                                const float* __restrict__ gamma, const float* __restrict__ beta,
                                const float* __restrict__ res, float* __restrict__ out,
                                int rows, int C, int doElu) {
    int i = blockIdx.x * blockDim.x + threadIdx.x;
    if (i >= rows * C) return;
    int c = i % C;
    float mean = sums[c] / (float)rows;
    float var  = sums[C + c] / (float)rows - mean * mean;
    float y = gamma[c] * (x[i] - mean) * rsqrtf(var + BN_EPS) + beta[c];
    if (doElu) y = (y > 0.f) ? y : expm1f(y);
    if (res)   y += res[i];
    out[i] = y;
}

// ---------------------------------------------------------------------------
// WMMA GEMM : Out[M,Nc] = A[M,K] @ W[K,Nc] + bias  (optionally ReLU)
// row-major; M % 16 == 0, K % 32 == 0, Nc % 16 == 0
// one wave -> one 16x16 tile; 4 waves / block cover 4 N-tiles
// A fragment per lane = two contiguous 8-float runs -> four b128 loads
// ---------------------------------------------------------------------------
__global__ __launch_bounds__(128)
void wmma_gemm_kernel(const float* __restrict__ A, const float* __restrict__ W,
                      const float* __restrict__ bias, float* __restrict__ Out,
                      int M, int K, int Nc, int doRelu) {
    const int lane  = threadIdx.x;                 // 0..31 (wave32)
    const int ntile = blockIdx.y * 4 + threadIdx.y;
    if (ntile * 16 >= Nc) return;                  // wave-uniform guard
    const int m0    = blockIdx.x * 16;
    const int n0    = ntile * 16;
    const int mrow  = m0 + (lane & 15);
    const int ncol  = n0 + (lane & 15);
    const int khalf = (lane < 16) ? 0 : 8;         // ISA 16-bit A/B lane-half K offset

    const f32x4* Ar = (const f32x4*)(A + (size_t)mrow * K);   // 16B-aligned (K % 4 == 0)

    v8f c = {};
    for (int k0 = 0; k0 < K; k0 += 32) {
        // ---- A fragment: a[2v+j] maps sequentially onto 16 contiguous floats
        //      (k0+khalf .. +7) then (k0+16+khalf .. +7)  -> 4x global_load_b128
        f32x4 a0 = Ar[(k0 + khalf) >> 2];
        f32x4 a1 = Ar[((k0 + khalf) >> 2) + 1];
        f32x4 a2 = Ar[(k0 + 16 + khalf) >> 2];
        f32x4 a3 = Ar[((k0 + 16 + khalf) >> 2) + 1];
        v16h a;
#pragma unroll
        for (int j = 0; j < 4; ++j) {
            a[j]      = (_Float16)a0[j];
            a[4 + j]  = (_Float16)a1[j];
            a[8 + j]  = (_Float16)a2[j];
            a[12 + j] = (_Float16)a3[j];
        }
        // ---- B fragment: per-lane strided, coalesced across lanes (ncol contiguous)
        v16h b;
#pragma unroll
        for (int v = 0; v < 8; ++v) {
            int k = k0 + ((v < 4) ? 0 : 16) + (v & 3) * 2 + khalf;
            b[2 * v]     = (_Float16)W[(size_t)k * Nc + ncol];
            b[2 * v + 1] = (_Float16)W[(size_t)(k + 1) * Nc + ncol];
        }
        c = __builtin_amdgcn_wmma_f32_16x16x32_f16(
                /*neg_a=*/false, a, /*neg_b=*/false, b,
                /*c_mod=*/(short)0, c, /*reuse_a=*/false, /*reuse_b=*/false);
    }
    const int   mb = m0 + ((lane < 16) ? 0 : 8);   // D layout: M = r + lanehalf*8
    const float bs = bias ? bias[ncol] : 0.f;
#pragma unroll
    for (int r = 0; r < 8; ++r) {
        float vo = c[r] + bs;
        if (doRelu) vo = fmaxf(vo, 0.f);
        Out[(size_t)(mb + r) * Nc + ncol] = vo;
    }
}

// ---------------------------------------------------------------------------
// edge kernels (wave-per-edge; Ctot = 32*CPL channels, CPL per lane)
// ---------------------------------------------------------------------------
__device__ inline void atomicMaxF(float* addr, float val) {
    if (val >= 0.f) atomicMax((int*)addr, __float_as_int(val));
    else            atomicMin((unsigned int*)addr, __float_as_uint(val));
}

__device__ inline void edge_endpoints(const ei_t* __restrict__ ei, int wid, ei_t& s, ei_t& d) {
    if (wid < NEDGES) { s = ei[wid]; d = ei[NEDGES + wid]; }
    else              { s = d = (ei_t)(wid - NEDGES); }
}

// score[e,h] = sum_c att[h,c] * leaky_relu(xl[src,h,c] + xr[dst,h,c]); seg-max into smax
template<int H, int CPL>
__global__ void edge_score_kernel(const float* __restrict__ xl, const float* __restrict__ xr,
                                  const float* __restrict__ att, const ei_t* __restrict__ ei,
                                  float* __restrict__ score, float* __restrict__ smax) {
    typedef typename VecOf<CPL>::type vec_t;
    const int wid  = (blockIdx.x * blockDim.x + threadIdx.x) >> 5;
    const int lane = threadIdx.x & 31;
    if (wid >= NETOT) return;                       // wave-uniform
    ei_t s, d; edge_endpoints(ei, wid, s, d);

    const int Ctot = 32 * CPL;
    const int base = lane * CPL;
    vec_t L = *(const vec_t*)(xl + (size_t)s * Ctot + base);  // one b128/b64 load
    vec_t R = *(const vec_t*)(xr + (size_t)d * Ctot + base);
    vec_t A = *(const vec_t*)(att + base);
    float part = 0.f;
#pragma unroll
    for (int j = 0; j < CPL; ++j) {
        float v = L[j] + R[j];
        v = (v > 0.f) ? v : v * LRELU_S;
        part += v * A[j];
    }
    const int lph = 32 / H;                         // lanes per head
#pragma unroll
    for (int off = lph >> 1; off > 0; off >>= 1)
        part += __shfl_xor(part, off, 32);
    if ((lane & (lph - 1)) == 0) {
        int h = lane / lph;
        score[(size_t)wid * H + h] = part;
        atomicMaxF(&smax[(size_t)d * H + h], part);
    }
}

// p = exp(score - smax[dst]); seg-sum into denom
__global__ void edge_exp_kernel(const ei_t* __restrict__ ei, const float* __restrict__ score,
                                const float* __restrict__ smax, float* __restrict__ p,
                                float* __restrict__ denom, int H) {
    int idx = blockIdx.x * blockDim.x + threadIdx.x;
    if (idx >= NETOT * H) return;
    int e = idx / H, h = idx - e * H;
    ei_t d = (e < NEDGES) ? ei[NEDGES + e] : (ei_t)(e - NEDGES);
    float v = expf(score[idx] - smax[(size_t)d * H + h]);
    p[idx] = v;
    atomicAdd(&denom[(size_t)d * H + h], v);
}

// agg[dst] += xl[src] * alpha  (alpha = p/denom[dst]); float atomics
template<int H, int CPL>
__global__ void edge_agg_kernel(const float* __restrict__ xl, const ei_t* __restrict__ ei,
                                const float* __restrict__ p, const float* __restrict__ denom,
                                float* __restrict__ agg) {
    typedef typename VecOf<CPL>::type vec_t;
    const int wid  = (blockIdx.x * blockDim.x + threadIdx.x) >> 5;
    const int lane = threadIdx.x & 31;
    if (wid >= NETOT) return;                       // wave-uniform
    ei_t s, d; edge_endpoints(ei, wid, s, d);

    const int Ctot = 32 * CPL;
    const int base = lane * CPL;
    const int C    = Ctot / H;
    const int h    = base / C;                      // all CPL channels share one head
    const float alpha = p[(size_t)wid * H + h] / denom[(size_t)d * H + h];
    vec_t L = *(const vec_t*)(xl + (size_t)s * Ctot + base);
#pragma unroll
    for (int j = 0; j < CPL; ++j)
        atomicAdd(&agg[(size_t)d * Ctot + base + j], L[j] * alpha);
}

// ---------------------------------------------------------------------------
// host-side helpers (kernel launches only — graph-capture safe)
// ---------------------------------------------------------------------------
static inline void launch_fill(float* p, float v, int n, hipStream_t s) {
    fill_f32_kernel<<<(n + 255) / 256, 256, 0, s>>>(p, v, n);
}
static inline void launch_gemm(const float* A, const float* W, const float* bias, float* Out,
                               int M, int K, int Nc, int relu, hipStream_t s) {
    dim3 blk(32, 4), grd(M / 16, (Nc / 16 + 3) / 4);
    wmma_gemm_kernel<<<grd, blk, 0, s>>>(A, W, bias, Out, M, K, Nc, relu);
}
static inline void launch_bn(const float* x, float* sums, const float* g, const float* b,
                             const float* res, float* out, int rows, int C, int elu,
                             hipStream_t s) {
    launch_fill(sums, 0.f, 2 * C, s);
    bn_stats_kernel<<<256, C, 0, s>>>(x, sums, rows, C);
    bn_apply_kernel<<<(rows * C + 255) / 256, 256, 0, s>>>(x, sums, g, b, res, out, rows, C, elu);
}
template<int H, int CPL>
static inline void launch_gat_edges(const float* xl, const float* xr, const float* att,
                                    const ei_t* ei, float* score, float* pbuf,
                                    float* smax, float* denom, float* agg, hipStream_t s) {
    const int Ctot = 32 * CPL;
    launch_fill(smax, -3.4028235e38f, NNODES * H, s);
    launch_fill(denom, 0.f, NNODES * H, s);
    launch_fill(agg, 0.f, NNODES * Ctot, s);
    const int edgeWaveBlocks = (NETOT * 32 + 255) / 256;
    edge_score_kernel<H, CPL><<<edgeWaveBlocks, 256, 0, s>>>(xl, xr, att, ei, score, smax);
    edge_exp_kernel<<<(NETOT * H + 255) / 256, 256, 0, s>>>(ei, score, smax, pbuf, denom, H);
    edge_agg_kernel<H, CPL><<<edgeWaveBlocks, 256, 0, s>>>(xl, ei, pbuf, denom, agg);
}

extern "C" void kernel_launch(void* const* d_in, const int* in_sizes, int n_in,
                              void* d_out, int out_size, void* d_ws, size_t ws_size,
                              hipStream_t stream) {
    // ---- inputs (setup_inputs dict order) ----
    const float* x         = (const float*)d_in[0];
    const ei_t*  ei        = (const ei_t*) d_in[1];
    const float* bnin_g    = (const float*)d_in[2];
    const float* bnin_b    = (const float*)d_in[3];
    const float* c0_Wl     = (const float*)d_in[4];
    const float* c0_bl     = (const float*)d_in[5];
    const float* c0_Wr     = (const float*)d_in[6];
    const float* c0_br     = (const float*)d_in[7];
    const float* c0_att    = (const float*)d_in[8];
    /* c0_bias d_in[9]: per-channel constant, cancelled exactly by following BN */
    const float* bn0_g     = (const float*)d_in[10];
    const float* bn0_b     = (const float*)d_in[11];
    const float* c1_Wl     = (const float*)d_in[12];
    const float* c1_bl     = (const float*)d_in[13];
    const float* c1_Wr     = (const float*)d_in[14];
    const float* c1_br     = (const float*)d_in[15];
    const float* c1_att    = (const float*)d_in[16];
    /* c1_bias d_in[17]: cancelled by BN */
    const float* bn1_g     = (const float*)d_in[18];
    const float* bn1_b     = (const float*)d_in[19];
    const float* c2_Wl     = (const float*)d_in[20];
    const float* c2_bl     = (const float*)d_in[21];
    const float* c2_Wr     = (const float*)d_in[22];
    const float* c2_br     = (const float*)d_in[23];
    const float* c2_att    = (const float*)d_in[24];
    const float* c2_bias   = (const float*)d_in[25];
    const float* cls_w1    = (const float*)d_in[26];
    const float* cls_b1    = (const float*)d_in[27];
    const float* cls_w2    = (const float*)d_in[28];
    const float* cls_b2    = (const float*)d_in[29];
    float* out = (float*)d_out;                      // [N, OUT] fp32

    // ---- workspace carve-out (256B-aligned blocks) ----
    char* w = (char*)d_ws;
    auto carve = [&](size_t nfloats) -> float* {
        float* p = (float*)w;
        w += (nfloats * sizeof(float) + 255) & ~(size_t)255;
        return p;
    };
    const size_t NF = (size_t)NNODES * CIN;          // 6.4M floats
    float* hA    = carve(NF);                        // features ping
    float* hB    = carve(NF);                        // features pong
    float* xl    = carve(NF);                        // source transform
    float* xr    = carve(NF);                        // target transform
    float* agg   = carve(NF);                        // segment-sum output
    float* score = carve((size_t)NETOT * NHEADS);
    float* pbuf  = carve((size_t)NETOT * NHEADS);
    float* smax  = carve((size_t)NNODES * NHEADS);
    float* denom = carve((size_t)NNODES * NHEADS);
    float* stats = carve(2 * CIN);
    float* t32   = carve((size_t)NNODES * CHID);     // classifier hidden

    const int M = NNODES, HH = NHEADS * CHID;        // HH == 128

    // ---- input batch norm: hA = bn(x) ----
    launch_bn(x, stats, bnin_g, bnin_b, /*res=*/nullptr, hA, M, CIN, /*elu=*/0, stream);

    // ---- GAT layer 0: hB = elu(bn(gatv2(hA))) + hA ----
    launch_gemm(hA, c0_Wl, c0_bl, xl, M, CIN, HH, 0, stream);
    launch_gemm(hA, c0_Wr, c0_br, xr, M, CIN, HH, 0, stream);
    launch_gat_edges<NHEADS, 4>(xl, xr, c0_att, ei, score, pbuf, smax, denom, agg, stream);
    launch_bn(agg, stats, bn0_g, bn0_b, /*res=*/hA, hB, M, HH, /*elu=*/1, stream);

    // ---- GAT layer 1: hA = elu(bn(gatv2(hB))) + hB ----
    launch_gemm(hB, c1_Wl, c1_bl, xl, M, HH, HH, 0, stream);
    launch_gemm(hB, c1_Wr, c1_br, xr, M, HH, HH, 0, stream);
    launch_gat_edges<NHEADS, 4>(xl, xr, c1_att, ei, score, pbuf, smax, denom, agg, stream);
    launch_bn(agg, stats, bn1_g, bn1_b, /*res=*/hB, hA, M, HH, /*elu=*/1, stream);

    // ---- GAT layer 2: heads=1, concat=False (mean of 1 head = identity), + bias ----
    launch_gemm(hA, c2_Wl, c2_bl, xl, M, HH, COUT, 0, stream);
    launch_gemm(hA, c2_Wr, c2_br, xr, M, HH, COUT, 0, stream);
    launch_gat_edges<1, 2>(xl, xr, c2_att, ei, score, pbuf, smax, denom, agg, stream);
    add_bias_kernel<<<(M * COUT + 255) / 256, 256, 0, stream>>>(agg, c2_bias, M, COUT);

    // ---- classifier MLP: out = relu(agg @ w1 + b1) @ w2 + b2 ----
    launch_gemm(agg, cls_w1, cls_b1, t32, M, COUT, CHID, /*relu=*/1, stream);
    launch_gemm(t32, cls_w2, cls_b2, out, M, CHID, COUT, /*relu=*/0, stream);

    (void)in_sizes; (void)n_in; (void)out_size; (void)ws_size;
}
